// GRUModule_13262859010311
// MI455X (gfx1250) — compile-verified, matrix operand order checked by reading, not run on hardware
//
#include <hip/hip_runtime.h>
#include <cstdint>
#include <cstddef>

// ---------------------------------------------------------------------------
// GRU forward for MI455X (gfx1250): bf16 WMMA GEMMs + cooperative scan with
// async-to-LDS staging of the recurrent state.
// ---------------------------------------------------------------------------

typedef __bf16 bf16_t;
typedef __attribute__((ext_vector_type(8)))  __bf16 v8bf;
typedef __attribute__((ext_vector_type(16))) __bf16 v16bf;
typedef __attribute__((ext_vector_type(8)))  float  v8f;
typedef __attribute__((ext_vector_type(4)))  float  v4f;
typedef __attribute__((ext_vector_type(4)))  int    v4i;

#define Bn   64
#define Tn   1024
#define INn  256
#define Hn   512
#define G3H  1536
#define OUTn 256
#define HPITCH 520   // LDS row pitch (elements) for h tile: avoids bank conflicts

#define AS1 __attribute__((address_space(1)))
#define AS3 __attribute__((address_space(3)))

#if __has_builtin(__builtin_amdgcn_global_load_async_to_lds_b128)
#define HAVE_ASYNC_LDS 1
#else
#define HAVE_ASYNC_LDS 0
#endif

// ---- helpers --------------------------------------------------------------

static __device__ __forceinline__ v8f wmma_bf16(v16bf a, v16bf b, v8f c) {
  // D = A(16x32 bf16) * B(32x16 bf16) + C(16x16 f32)
  return __builtin_amdgcn_wmma_f32_16x16x32_bf16(
      /*neg_a=*/false, a, /*neg_b=*/false, b,
      /*c_mod=*/(short)0, c, /*reuse_a=*/false, /*reuse_b=*/false);
}

// Load one lane's 16 bf16 A/B-operand values from a row-major [rows x K]
// matrix (global or LDS). Per ISA 7.12.2: lane L (hi = L>>4) holds
// K = {hi*8..hi*8+7} in slots 0..7 and K = {16+hi*8..+7} in slots 8..15 —
// two contiguous 16-byte chunks -> two b128 loads (global or ds).
static __device__ __forceinline__ v16bf ld_tile(const bf16_t* row, int kk, int hi) {
  v8bf lo = *(const v8bf*)(row + kk + hi * 8);
  v8bf hh = *(const v8bf*)(row + kk + 16 + hi * 8);
  return __builtin_shufflevector(lo, hh, 0, 1, 2, 3, 4, 5, 6, 7,
                                 8, 9, 10, 11, 12, 13, 14, 15);
}

// Copy 16 bytes global -> LDS. Async (GLOBAL_LOAD_ASYNC_TO_LDS_B128, tracked
// by ASYNCcnt) when the toolchain exposes the builtin; else sync load+store.
static __device__ __forceinline__ void copy16_g2l(const bf16_t* g, bf16_t* l) {
#if HAVE_ASYNC_LDS
  __builtin_amdgcn_global_load_async_to_lds_b128(
      (AS1 v4i*)g, (AS3 v4i*)l, /*offset=*/0, /*cpol=*/0);
#else
  *(v8bf*)l = *(const v8bf*)g;  // global_load_b128 + ds_store_b128
#endif
}

static __device__ __forceinline__ void wait_g2l() {
#if HAVE_ASYNC_LDS
#if __has_builtin(__builtin_amdgcn_s_wait_asynccnt)
  __builtin_amdgcn_s_wait_asynccnt(0);
#else
  asm volatile("s_wait_asynccnt 0" ::: "memory");
#endif
#endif
}

// Load 8 consecutive f32 and round to bf16 (RNE via __bf16 cast).
static __device__ __forceinline__ v8bf cvt8(const float* p) {
  v4f a = *(const v4f*)p;
  v4f b = *(const v4f*)(p + 4);
  v8bf r;
#pragma unroll
  for (int i = 0; i < 4; ++i) { r[i] = (bf16_t)a[i]; r[i + 4] = (bf16_t)b[i]; }
  return r;
}

static __device__ __forceinline__ float sigmoidf_(float x) {
  return 1.0f / (1.0f + __expf(-x));
}
static __device__ __forceinline__ float tanhf_(float x) {
  float e = __expf(2.0f * x);
  return (e - 1.0f) / (e + 1.0f);
}

// ---- kernel 1: weight conversion + zero-init ------------------------------

__global__ void prep_kernel(const float* __restrict__ w_ih,
                            const float* __restrict__ w_hh,
                            const float* __restrict__ fc_w,
                            bf16_t* __restrict__ wih_bf,
                            bf16_t* __restrict__ whh_bf,
                            bf16_t* __restrict__ fcw_bf,
                            bf16_t* __restrict__ hbuf,
                            float* __restrict__ h0_out,
                            unsigned* __restrict__ bar) {
  int i = blockIdx.x * blockDim.x + threadIdx.x;
  const int N0 = G3H * INn;        // w_ih
  const int N1 = N0 + G3H * Hn;    // w_hh
  const int N2 = N1 + OUTn * Hn;   // fc_w
  const int N3 = N2 + 2 * Bn * Hn; // h ping-pong buffers (zero)
  const int N4 = N3 + Bn * Hn;     // h0 output region (zero)
  const int N5 = N4 + 16;          // barrier counters
  if (i < N0)       wih_bf[i]      = (bf16_t)w_ih[i];
  else if (i < N1)  whh_bf[i - N0] = (bf16_t)w_hh[i - N0];
  else if (i < N2)  fcw_bf[i - N1] = (bf16_t)fc_w[i - N1];
  else if (i < N3)  hbuf[i - N2]   = (bf16_t)0.0f;
  else if (i < N4)  h0_out[i - N3] = 0.0f;
  else if (i < N5)  bar[i - N4]    = 0u;
}

// ---- kernel 2: gi_all[t][b][g] = x[b,t,:] . w_ih[g,:] + bias_ih[g] --------
// grid = (12 n-chunks, T). One 16x16 output tile per wave, 4 m-tiles.

__global__ void gi_kernel(const float* __restrict__ x,
                          const bf16_t* __restrict__ wih_bf,
                          const float* __restrict__ bias_ih,
                          bf16_t* __restrict__ gi) {
  const int t    = blockIdx.y;
  const int wave = threadIdx.x >> 5;
  const int lane = threadIdx.x & 31;
  const int n    = lane & 15;
  const int hi   = lane >> 4;
  const int gcol0 = (blockIdx.x * 8 + wave) * 16;
  const int jcol  = gcol0 + n;
  const bf16_t* brow = wih_bf + (size_t)jcol * INn;
  const float bi = bias_ih[jcol];
  bf16_t* git = gi + (size_t)t * Bn * G3H;

#pragma unroll 1
  for (int m0 = 0; m0 < Bn / 16; ++m0) {
    const float* arow = x + (size_t)(m0 * 16 + n) * Tn * INn + (size_t)t * INn;
    v8f acc = {};
#pragma unroll
    for (int kk = 0; kk < INn; kk += 32) {
      v16bf a = __builtin_shufflevector(cvt8(arow + kk + hi * 8),
                                        cvt8(arow + kk + 16 + hi * 8),
                                        0, 1, 2, 3, 4, 5, 6, 7,
                                        8, 9, 10, 11, 12, 13, 14, 15);
      v16bf b = ld_tile(brow, kk, hi);
      acc = wmma_bf16(a, b, acc);
    }
#pragma unroll
    for (int j = 0; j < 8; ++j) {
      int brw = m0 * 16 + j + 8 * hi; // batch row of this D element
      git[(size_t)brw * G3H + jcol] = (bf16_t)(acc[j] + bi);
    }
  }
}

// ---- kernel 3: recurrent scan ---------------------------------------------
// 16 blocks = 4 independent batch groups x 4 column-parts; 8 waves/block,
// one 16x16 h-tile per wave (3 gate accumulators). Each step: stage the
// group's 16x512 bf16 h tile into LDS (async-to-LDS when available), run the
// K-loop with A from LDS (ds_load_b128) and W_hh from global, apply gates,
// ping-pong h, and sync the 4 blocks of the group with a monotonic atomic
// barrier.

__global__ void scan_kernel(const bf16_t* __restrict__ gi,
                            const bf16_t* __restrict__ whh_bf,
                            const float* __restrict__ bias_hh,
                            bf16_t* __restrict__ hbuf,
                            unsigned* __restrict__ bar,
                            float* __restrict__ outs,
                            float* __restrict__ hn_out) {
  __shared__ bf16_t lds_h[16 * HPITCH];  // padded pitch: rows 4 banks apart

  const int grp  = blockIdx.x >> 2;   // batch group 0..3 (16 rows each)
  const int part = blockIdx.x & 3;    // column part 0..3
  const int wave = threadIdx.x >> 5;
  const int lane = threadIdx.x & 31;
  const int n    = lane & 15;
  const int hi   = lane >> 4;
  const int j0   = (part * 8 + wave) * 16; // h-column tile base
  const int jcol = j0 + n;

  const bf16_t* wr = whh_bf + (size_t)(0 * Hn + jcol) * Hn;
  const bf16_t* wz = whh_bf + (size_t)(1 * Hn + jcol) * Hn;
  const bf16_t* wn = whh_bf + (size_t)(2 * Hn + jcol) * Hn;
  const float bhr = bias_hh[0 * Hn + jcol];
  const float bhz = bias_hh[1 * Hn + jcol];
  const float bhn = bias_hh[2 * Hn + jcol];
  unsigned* cnt = bar + grp;

  for (int t = 0; t < Tn; ++t) {
    const bf16_t* hc = hbuf + (size_t)(t & 1) * Bn * Hn;
    bf16_t*       hx = hbuf + (size_t)((t + 1) & 1) * Bn * Hn;

    // ---- stage h[grp] (16x512 bf16 = 1024 x 16B chunks) into LDS ----
    const bf16_t* hsrc = hc + (size_t)grp * 16 * Hn;
#pragma unroll
    for (int it = 0; it < 4; ++it) {
      const int c   = it * 256 + threadIdx.x; // chunk id 0..1023
      const int row = c >> 6;                 // 64 chunks per row
      const int col = (c & 63) * 8;           // element offset in row
      copy16_g2l(hsrc + (size_t)row * Hn + col, lds_h + row * HPITCH + col);
    }
    wait_g2l();
    __syncthreads();

    // ---- K-loop: 3 gate tiles, A from LDS, B (weights) from global ----
    const bf16_t* arow = lds_h + n * HPITCH; // A-operand row m = lane&15
    v8f ar = {}, az = {}, an = {};
#pragma unroll 4
    for (int kk = 0; kk < Hn; kk += 32) {
      if (kk + 32 < Hn) { // prefetch next weight K-chunk (global_prefetch_b8)
        __builtin_prefetch(wr + kk + 32, 0, 1);
        __builtin_prefetch(wz + kk + 32, 0, 1);
        __builtin_prefetch(wn + kk + 32, 0, 1);
      }
      v16bf a  = ld_tile(arow, kk, hi);  // ds_load_b128 x2
      v16bf br = ld_tile(wr, kk, hi);
      v16bf bz = ld_tile(wz, kk, hi);
      v16bf bn = ld_tile(wn, kk, hi);
      ar = wmma_bf16(a, br, ar);
      az = wmma_bf16(a, bz, az);
      an = wmma_bf16(a, bn, an);
    }

    // ---- gate fusion + state update (D layout: m = j+8*hi, n = lane&15) ----
    const bf16_t* git = gi + (size_t)t * Bn * G3H;
#pragma unroll
    for (int j = 0; j < 8; ++j) {
      const int ml = j + 8 * hi;       // local batch row
      const int b  = grp * 16 + ml;    // global batch row
      const bf16_t* gib = git + (size_t)b * G3H + jcol;
      float r  = sigmoidf_((float)gib[0 * Hn] + ar[j] + bhr);
      float z  = sigmoidf_((float)gib[1 * Hn] + az[j] + bhz);
      float nn = tanhf_((float)gib[2 * Hn] + r * (an[j] + bhn));
      float hold = (float)lds_h[ml * HPITCH + jcol];
      float hnew = (1.0f - z) * nn + z * hold;
      hx[(size_t)b * Hn + jcol] = (bf16_t)hnew;
      outs[(size_t)t * Bn * Hn + (size_t)b * Hn + jcol] = hnew;
      if (t == Tn - 1) hn_out[(size_t)b * Hn + jcol] = hnew;
    }

    // ---- barrier among the 4 blocks of this batch group ----
    __syncthreads();
    if (threadIdx.x == 0) {
      __threadfence();
      atomicAdd(cnt, 1u);
      const unsigned target = 4u * (unsigned)(t + 1);
      while (atomicAdd(cnt, 0u) < target) { __builtin_amdgcn_s_sleep(1); }
    }
    __syncthreads();
  }
}

// ---- kernel 4: out[b,o] = h_last[b,:] . fc_w[o,:] + fc_b[o] ---------------
// Final h lives in hbuf[0] (T=1024 is even). Single block, 8 waves.

__global__ void fc_kernel(const bf16_t* __restrict__ hbuf,
                          const bf16_t* __restrict__ fcw_bf,
                          const float* __restrict__ fc_b,
                          float* __restrict__ out) {
  const int wave = threadIdx.x >> 5;
  const int lane = threadIdx.x & 31;
  const int n    = lane & 15;
  const int hi   = lane >> 4;

#pragma unroll 1
  for (int nt = wave; nt < OUTn / 16; nt += 8) {
    const int ocol = nt * 16 + n;
    const bf16_t* brow = fcw_bf + (size_t)ocol * Hn;
    const float bb = fc_b[ocol];
#pragma unroll 1
    for (int m0 = 0; m0 < Bn / 16; ++m0) {
      const bf16_t* arow = hbuf + (size_t)(m0 * 16 + n) * Hn;
      v8f acc = {};
#pragma unroll 4
      for (int kk = 0; kk < Hn; kk += 32) {
        v16bf a = ld_tile(arow, kk, hi);
        v16bf b = ld_tile(brow, kk, hi);
        acc = wmma_bf16(a, b, acc);
      }
#pragma unroll
      for (int j = 0; j < 8; ++j) {
        int brw = m0 * 16 + j + 8 * hi;
        out[(size_t)brw * OUTn + ocol] = acc[j] + bb;
      }
    }
  }
}

// ---- host side ------------------------------------------------------------

extern "C" void kernel_launch(void* const* d_in, const int* in_sizes, int n_in,
                              void* d_out, int out_size, void* d_ws, size_t ws_size,
                              hipStream_t stream) {
  (void)in_sizes; (void)n_in; (void)out_size; (void)ws_size;

  const float* x       = (const float*)d_in[0];
  const float* w_ih    = (const float*)d_in[1];
  const float* w_hh    = (const float*)d_in[2];
  const float* bias_ih = (const float*)d_in[3];
  const float* bias_hh = (const float*)d_in[4];
  const float* fc_w    = (const float*)d_in[5];
  const float* fc_b    = (const float*)d_in[6];
  float* out = (float*)d_out;

  // workspace carve-up (bytes)
  char* ws = (char*)d_ws;
  size_t off = 0;
  bf16_t* gi     = (bf16_t*)(ws + off); off += (size_t)Tn * Bn * G3H * 2; // 201 MB
  bf16_t* wih_bf = (bf16_t*)(ws + off); off += (size_t)G3H * INn * 2;
  bf16_t* whh_bf = (bf16_t*)(ws + off); off += (size_t)G3H * Hn * 2;
  bf16_t* fcw_bf = (bf16_t*)(ws + off); off += (size_t)OUTn * Hn * 2;
  bf16_t* hbuf   = (bf16_t*)(ws + off); off += (size_t)2 * Bn * Hn * 2;
  unsigned* bar  = (unsigned*)(ws + off); off += 64;

  // d_out carve-up: (outs[T,B,H], out[B,OUT], hn[B,H], h0[1,B,H])
  float* outs   = out;
  float* fcout  = outs + (size_t)Tn * Bn * Hn;
  float* hn_out = fcout + (size_t)Bn * OUTn;
  float* h0_out = hn_out + (size_t)Bn * Hn;

  const int prepN = G3H * INn + G3H * Hn + OUTn * Hn + 2 * Bn * Hn + Bn * Hn + 16;
  prep_kernel<<<(prepN + 255) / 256, 256, 0, stream>>>(
      w_ih, w_hh, fc_w, wih_bf, whh_bf, fcw_bf, hbuf, h0_out, bar);

  gi_kernel<<<dim3(G3H / 128, Tn), 256, 0, stream>>>(x, wih_bf, bias_ih, gi);

  scan_kernel<<<16, 256, 0, stream>>>(gi, whh_bf, bias_hh, hbuf, bar, outs, hn_out);

  fc_kernel<<<1, 256, 0, stream>>>(hbuf, fcw_bf, fc_b, fcout);
}
